// AttentionBlock_41996190220332
// MI455X (gfx1250) — matrix operand order repaired and checked by reading, not compiled, not run on hardware
//
#include <hip/hip_runtime.h>

// ---------------------------------------------------------------------------
// Types / constants
// ---------------------------------------------------------------------------
typedef __bf16 bf16;
typedef __attribute__((ext_vector_type(16))) __bf16 bf16x16;
typedef __attribute__((ext_vector_type(8)))  __bf16 bf16x8;
typedef __attribute__((ext_vector_type(8)))  float  f32x8;

#define DEVFN __device__ __forceinline__

constexpr int NB    = 4;
constexpr int SEQ   = 2048;
constexpr int DM    = 1024;
constexpr int NH    = 16;
constexpr int DH    = 64;       // DM / NH
constexpr int MROWS = NB * SEQ; // 8192

#define NEG_INF (-__builtin_inff())

DEVFN bf16 f2bf(float f) {
  unsigned u = __builtin_bit_cast(unsigned, f);
  u += 0x7FFFu + ((u >> 16) & 1u);          // round-to-nearest-even
  unsigned short h = (unsigned short)(u >> 16);
  return __builtin_bit_cast(bf16, h);
}
DEVFN float bf2f(bf16 b) {
  unsigned short h = __builtin_bit_cast(unsigned short, b);
  unsigned u = ((unsigned)h) << 16;
  return __builtin_bit_cast(float, u);
}
DEVFN f32x8 wmma_bf16(bf16x16 a, bf16x16 b, f32x8 c) {
  return __builtin_amdgcn_wmma_f32_16x16x32_bf16(false, a, false, b,
                                                 (short)0, c, false, false);
}
DEVFN bf16x16 cat8(bf16x8 lo, bf16x8 hi) {
  return __builtin_shufflevector(lo, hi, 0, 1, 2, 3, 4, 5, 6, 7,
                                 8, 9, 10, 11, 12, 13, 14, 15);
}

// ---------------------------------------------------------------------------
// Weight cast fp32 -> bf16
// ---------------------------------------------------------------------------
__global__ __launch_bounds__(256)
void cast_w(const float* __restrict__ w, bf16* __restrict__ o) {
  size_t i = (size_t)blockIdx.x * 256 + threadIdx.x;
  o[i] = f2bf(w[i]);
}

// ---------------------------------------------------------------------------
// LayerNorm (per row of DM=1024) + cast to bf16
// ---------------------------------------------------------------------------
__global__ __launch_bounds__(128)
void ln_cast(const float* __restrict__ x, const float* __restrict__ g,
             const float* __restrict__ bta, bf16* __restrict__ out) {
  const int row = blockIdx.x;
  const int t   = threadIdx.x;
  const float* xr = x + (size_t)row * DM;
  float v[8];
  float s = 0.f, s2 = 0.f;
#pragma unroll
  for (int i = 0; i < 8; ++i) {
    v[i] = xr[t + i * 128];
    s  += v[i];
    s2 += v[i] * v[i];
  }
#pragma unroll
  for (int off = 1; off < 32; off <<= 1) {
    s  += __shfl_xor(s,  off, 32);
    s2 += __shfl_xor(s2, off, 32);
  }
  __shared__ float red[8];
  if ((t & 31) == 0) { red[(t >> 5) * 2] = s; red[(t >> 5) * 2 + 1] = s2; }
  __syncthreads();
  s  = red[0] + red[2] + red[4] + red[6];
  s2 = red[1] + red[3] + red[5] + red[7];
  const float mu   = s * (1.f / (float)DM);
  const float var  = s2 * (1.f / (float)DM) - mu * mu;
  const float rstd = rsqrtf(var + 1e-5f);
#pragma unroll
  for (int i = 0; i < 8; ++i) {
    const int c = t + i * 128;
    out[(size_t)row * DM + c] = f2bf((v[i] - mu) * rstd * g[c] + bta[c]);
  }
}

// ---------------------------------------------------------------------------
// WMMA GEMM:  C[M,N] = A[M,K] @ W[N,K]^T   (nn.Linear), K = N = DM, M = MROWS
// block 256 thr = 8 waves, tile 128x128, per-wave 32x64 (2x4 wmma tiles):
// 8 WMMAs per 32-wide k-step against 12 b128 loads (1.5 loads/WMMA).
// ---------------------------------------------------------------------------
enum { EPI_F32 = 0, EPI_BF16 = 1, EPI_QKV = 2, EPI_VT = 3, EPI_RESID = 4 };

template <int EPI>
__global__ __launch_bounds__(256)
void gemm_bf16(const bf16* __restrict__ A, const bf16* __restrict__ W,
               void* __restrict__ outp, const float* __restrict__ resid) {
  constexpr int K = DM;
  const int lane = threadIdx.x & 31;
  const int wid  = threadIdx.x >> 5;
  const int l16  = lane & 15;
  const int half = lane >> 4;
  const int wm   = wid & 3;   // 4 waves along M
  const int wn   = wid >> 2;  // 2 waves along N
  const int m0   = blockIdx.x * 128 + wm * 32;
  const int n0   = blockIdx.y * 128 + wn * 64;

  f32x8 acc[2][4] = {};

  for (int k = 0; k < K; k += 32) {
    if (k + 32 < K) {  // speculative prefetch of next k-slab
      __builtin_prefetch(A + (size_t)(m0 + l16) * K + k + 32, 0, 1);
      __builtin_prefetch(W + (size_t)(n0 + l16) * K + k + 32, 0, 1);
    }
    bf16x16 af[2], bfg[4];
#pragma unroll
    for (int i = 0; i < 2; ++i) {
      const bf16* ap = A + (size_t)(m0 + i * 16 + l16) * K + k + half * 16;
      af[i] = cat8(*(const bf16x8*)(ap), *(const bf16x8*)(ap + 8));
    }
#pragma unroll
    for (int j = 0; j < 4; ++j) {
      const bf16* bp = W + (size_t)(n0 + j * 16 + l16) * K + k + half * 16;
      bfg[j] = *(const bf16x16*)(bp);
    }
#pragma unroll
    for (int i = 0; i < 2; ++i)
#pragma unroll
      for (int j = 0; j < 4; ++j)
        acc[i][j] = wmma_bf16(af[i], bfg[j], acc[i][j]);
  }

#pragma unroll
  for (int i = 0; i < 2; ++i) {
#pragma unroll
    for (int j = 0; j < 4; ++j) {
#pragma unroll
      for (int r = 0; r < 8; ++r) {
        const int row = m0 + i * 16 + half * 8 + r;   // C layout: M = r + 8*half
        const int col = n0 + j * 16 + l16;            // N = lane&15
        const float v = acc[i][j][r];
        if constexpr (EPI == EPI_F32) {
          ((float*)outp)[(size_t)row * DM + col] = v;
        } else if constexpr (EPI == EPI_BF16) {
          ((bf16*)outp)[(size_t)row * DM + col] = f2bf(v);
        } else if constexpr (EPI == EPI_QKV) {
          const int b = row / SEQ, s = row % SEQ;
          const int hh = col / DH, dh = col % DH;
          ((bf16*)outp)[(((size_t)(b * NH + hh) * SEQ) + s) * DH + dh] = f2bf(v);
        } else if constexpr (EPI == EPI_VT) {
          const int b = row / SEQ, s = row % SEQ;
          const int hh = col / DH, dh = col % DH;
          ((bf16*)outp)[(((size_t)(b * NH + hh) * DH) + dh) * SEQ + s] = f2bf(v);
        } else {  // EPI_RESID
          ((float*)outp)[(size_t)row * DM + col] =
              resid[(size_t)row * DM + col] + v;
        }
      }
    }
  }
}

// ---------------------------------------------------------------------------
// Flash attention with ALiBi + causal mask.
// q,k: (B,H,S,DH) bf16.  vt: (B,H,DH,S) bf16.  o: (B,S,DM) bf16.
// 1 wave per 16-query tile; 4 waves/block; online softmax.
//  - 1/sqrt(DH) folded into Q fragments (exact: bf16 * 2^-3)
//  - ALiBi bias carried incrementally in float regs (no per-block cvt/mul)
//  - softmax denominator accumulated by WMMA via a ones-column B fragment
//    (removes 32 ds_bpermute per key block; numerator/denominator use the
//    identical bf16 P values, so they stay consistent)
// ---------------------------------------------------------------------------
__global__ __launch_bounds__(128)
void attn_kernel(const bf16* __restrict__ q, const bf16* __restrict__ k,
                 const bf16* __restrict__ vt, bf16* __restrict__ o) {
  const int lane = threadIdx.x & 31;
  const int wid  = threadIdx.x >> 5;
  const int l16  = lane & 15;
  const int half = lane >> 4;
  const int bh   = blockIdx.y;          // b*NH + h
  const int b    = bh >> 4;
  const int hh   = bh & 15;
  const int q0   = (blockIdx.x * 4 + wid) * 16;
  const float slope = exp2f(-0.5f * (float)(hh + 1));  // 2^(-8/H)^(h+1)

  const bf16* qb = q  + (size_t)bh * SEQ * DH;
  const bf16* kb = k  + (size_t)bh * SEQ * DH;
  const bf16* vb = vt + (size_t)bh * DH * SEQ;

  // Q fragments (A-layout), pre-scaled by 1/sqrt(DH) = 2^-3 (exact in bf16)
  bf16x16 aQ[2];
#pragma unroll
  for (int d = 0; d < 2; ++d) {
    const bf16* p = qb + (size_t)(q0 + l16) * DH + d * 32 + half * 16;
    bf16x8 lo = *(const bf16x8*)(p);
    bf16x8 hi = *(const bf16x8*)(p + 8);
#pragma unroll
    for (int e = 0; e < 8; ++e) {
      aQ[d][e]     = f2bf(bf2f(lo[e]) * 0.125f);
      aQ[d][8 + e] = f2bf(bf2f(hi[e]) * 0.125f);
    }
  }

  // ones-column B fragment: B[k][0] = 1, other columns 0
  const bf16 kOne  = f2bf(1.f);
  const bf16 kZero = __builtin_bit_cast(bf16, (unsigned short)0);
  bf16x16 bOnes;
#pragma unroll
  for (int e = 0; e < 16; ++e) bOnes[e] = (l16 == 0) ? kOne : kZero;

  // o_acc[0..3] = output tiles; o_acc[4] column 0 = running softmax denominator
  f32x8 o_acc[5] = {};
  float rmax[8];
  float ab[2][8];   // running ALiBi bias: (kk - qq) * slope
#pragma unroll
  for (int r = 0; r < 8; ++r) {
    rmax[r] = NEG_INF;
    const int qq = q0 + half * 8 + r;
#pragma unroll
    for (int j = 0; j < 2; ++j)
      ab[j][r] = (float)(j * 16 + l16 - qq) * slope;
  }
  const float ab_step = 32.f * slope;

  __shared__ __align__(16) bf16 pshare[4][16][32];  // per-wave P tile
  bf16 (*pw)[32] = pshare[wid];

  const int nkb = (q0 + 16 + 31) / 32;  // 32-key blocks covering keys<=q0+15

  for (int kbI = 0; kbI < nkb; ++kbI) {
    const int kbase = kbI * 32;

    // scores tile 16q x 32k  (two 16x16 C tiles)
    f32x8 sc[2];
#pragma unroll
    for (int j = 0; j < 2; ++j) {
      f32x8 a = {};
#pragma unroll
      for (int d = 0; d < 2; ++d) {
        bf16x16 bK = *(const bf16x16*)(
            kb + (size_t)(kbase + j * 16 + l16) * DH + d * 32 + half * 16);
        a = wmma_bf16(aQ[d], bK, a);
      }
      sc[j] = a;
    }

    // ALiBi + causal; per-row block max
    float tmax[8];
#pragma unroll
    for (int r = 0; r < 8; ++r) {
      const int qq = q0 + half * 8 + r;
      float m2 = NEG_INF;
#pragma unroll
      for (int j = 0; j < 2; ++j) {
        const int kk = kbase + j * 16 + l16;
        float vv = sc[j][r] + ab[j][r];
        vv = (kk > qq) ? NEG_INF : vv;   // exact int compare for the mask
        sc[j][r] = vv;
        m2 = fmaxf(m2, vv);
        ab[j][r] += ab_step;
      }
      tmax[r] = m2;
    }
#pragma unroll
    for (int r = 0; r < 8; ++r) {
      float m2 = tmax[r];
#pragma unroll
      for (int off = 1; off < 16; off <<= 1)
        m2 = fmaxf(m2, __shfl_xor(m2, off, 32));
      tmax[r] = m2;
    }

    float rescale[8];
#pragma unroll
    for (int r = 0; r < 8; ++r) {
      const float nm = fmaxf(rmax[r], tmax[r]);
      rescale[r] = __expf(rmax[r] - nm);
      rmax[r] = nm;
    }

    // probabilities -> LDS (C-layout -> memory [m][k])
#pragma unroll
    for (int r = 0; r < 8; ++r) {
      const int mrow = half * 8 + r;
#pragma unroll
      for (int j = 0; j < 2; ++j)
        pw[mrow][j * 16 + l16] = f2bf(__expf(sc[j][r] - rmax[r]));
    }

    // rescale running output tiles + denominator column
#pragma unroll
    for (int jj = 0; jj < 5; ++jj)
#pragma unroll
      for (int r = 0; r < 8; ++r)
        o_acc[jj][r] *= rescale[r];

    // reload P as an A-fragment (wave-private region; order own DS ops)
    __asm__ volatile("s_wait_dscnt 0" ::: "memory");
    bf16x16 aP;
    {
      const bf16* pp = &pw[l16][half * 16];
#pragma unroll
      for (int e = 0; e < 8; ++e) { aP[e] = pp[e]; aP[8 + e] = pp[8 + e]; }
    }

    // O += P @ V  (contraction over 32 keys, 4 d-tiles of 16)
#pragma unroll
    for (int jj = 0; jj < 4; ++jj) {
      bf16x16 bV = *(const bf16x16*)(
          vb + (size_t)(jj * 16 + l16) * SEQ + kbase + half * 16);
      o_acc[jj] = wmma_bf16(aP, bV, o_acc[jj]);
    }
    // denominator: row-sum of P via ones column
    o_acc[4] = wmma_bf16(aP, bOnes, o_acc[4]);
  }

  // broadcast denominator (col 0 of tile 4 lives in lane l16==0 of each half)
  float rdiv[8];
#pragma unroll
  for (int r = 0; r < 8; ++r)
    rdiv[r] = 1.f / __shfl(o_acc[4][r], half * 16, 32);

  // normalize + store to (B,S,DM) bf16
#pragma unroll
  for (int jj = 0; jj < 4; ++jj) {
#pragma unroll
    for (int r = 0; r < 8; ++r) {
      const int s    = q0 + half * 8 + r;
      const int dcol = hh * DH + jj * 16 + l16;
      o[((size_t)(b * SEQ + s)) * DM + dcol] = f2bf(o_acc[jj][r] * rdiv[r]);
    }
  }
}

// ---------------------------------------------------------------------------
// SwiGLU gate: g = silu(z1) * z3
// ---------------------------------------------------------------------------
__global__ __launch_bounds__(256)
void silu_mul(const bf16* __restrict__ z1, const bf16* __restrict__ z3,
              bf16* __restrict__ g) {
  size_t i = (size_t)blockIdx.x * 256 + threadIdx.x;
  const float a = bf2f(z1[i]);
  const float c = bf2f(z3[i]);
  const float s = a / (1.f + __expf(-a));
  g[i] = f2bf(s * c);
}

// ---------------------------------------------------------------------------
// Host-side orchestration
// ---------------------------------------------------------------------------
extern "C" void kernel_launch(void* const* d_in, const int* in_sizes, int n_in,
                              void* d_out, int out_size, void* d_ws,
                              size_t ws_size, hipStream_t stream) {
  (void)in_sizes; (void)n_in; (void)out_size; (void)ws_size;
  const float* x     = (const float*)d_in[0];
  const float* wq    = (const float*)d_in[1];
  const float* wk    = (const float*)d_in[2];
  const float* wv    = (const float*)d_in[3];
  const float* wo    = (const float*)d_in[4];
  const float* ln1_g = (const float*)d_in[5];
  const float* ln1_b = (const float*)d_in[6];
  const float* ln2_g = (const float*)d_in[7];
  const float* ln2_b = (const float*)d_in[8];
  const float* w1    = (const float*)d_in[9];
  const float* w2    = (const float*)d_in[10];
  const float* w3    = (const float*)d_in[11];
  float* out = (float*)d_out;

  char* ws = (char*)d_ws;
  const size_t WSZ = (size_t)DM * DM * sizeof(bf16);       // 2 MB per weight
  bf16* wq_b = (bf16*)(ws + 0 * WSZ);
  bf16* wk_b = (bf16*)(ws + 1 * WSZ);
  bf16* wv_b = (bf16*)(ws + 2 * WSZ);
  bf16* wo_b = (bf16*)(ws + 3 * WSZ);
  bf16* w1_b = (bf16*)(ws + 4 * WSZ);
  bf16* w2_b = (bf16*)(ws + 5 * WSZ);
  bf16* w3_b = (bf16*)(ws + 6 * WSZ);
  const size_t ASZ = (size_t)MROWS * DM * sizeof(bf16);    // 16 MB each
  char* p = ws + 7 * WSZ;
  bf16* h_b = (bf16*)p; p += ASZ;  // LN output (reused for LN2 output)
  bf16* q_b = (bf16*)p; p += ASZ;  // Q (B,H,S,Dh)    -> later z1
  bf16* k_b = (bf16*)p; p += ASZ;  // K (B,H,S,Dh)    -> later z3
  bf16* v_b = (bf16*)p; p += ASZ;  // V^T (B,H,Dh,S)  -> later gate
  bf16* o_b = (bf16*)p;            // attention out (B,S,DM)

  const int nW = DM * DM / 256;                 // 4096 blocks
  cast_w<<<nW, 256, 0, stream>>>(wq, wq_b);
  cast_w<<<nW, 256, 0, stream>>>(wk, wk_b);
  cast_w<<<nW, 256, 0, stream>>>(wv, wv_b);
  cast_w<<<nW, 256, 0, stream>>>(wo, wo_b);
  cast_w<<<nW, 256, 0, stream>>>(w1, w1_b);
  cast_w<<<nW, 256, 0, stream>>>(w2, w2_b);
  cast_w<<<nW, 256, 0, stream>>>(w3, w3_b);

  const dim3 gG(MROWS / 128, DM / 128);         // 64 x 8 blocks
  const dim3 bG(256);

  // --- attention half ---
  ln_cast<<<MROWS, 128, 0, stream>>>(x, ln1_g, ln1_b, h_b);
  gemm_bf16<EPI_QKV><<<gG, bG, 0, stream>>>(h_b, wq_b, q_b, nullptr);
  gemm_bf16<EPI_QKV><<<gG, bG, 0, stream>>>(h_b, wk_b, k_b, nullptr);
  gemm_bf16<EPI_VT ><<<gG, bG, 0, stream>>>(h_b, wv_b, v_b, nullptr);
  attn_kernel<<<dim3(SEQ / 64, NB * NH), 128, 0, stream>>>(q_b, k_b, v_b, o_b);
  gemm_bf16<EPI_RESID><<<gG, bG, 0, stream>>>(o_b, wo_b, out, x);  // out = x + o@wo^T

  // --- SwiGLU FFN half ---
  ln_cast<<<MROWS, 128, 0, stream>>>(out, ln2_g, ln2_b, h_b);
  gemm_bf16<EPI_BF16><<<gG, bG, 0, stream>>>(h_b, w1_b, q_b, nullptr);   // z1
  gemm_bf16<EPI_BF16><<<gG, bG, 0, stream>>>(h_b, w3_b, k_b, nullptr);   // z3
  silu_mul<<<(size_t)MROWS * DM / 256, 256, 0, stream>>>(q_b, k_b, v_b); // gate
  gemm_bf16<EPI_RESID><<<gG, bG, 0, stream>>>(v_b, w2_b, out, out);      // out += g@w2^T
}
